// NonLocalBlock_75711683494170
// MI455X (gfx1250) — matrix operand order repaired and checked by reading, not compiled, not run on hardware
//
#include <hip/hip_runtime.h>
#include <hip/hip_bf16.h>

typedef __bf16 bf16_t;
typedef __attribute__((ext_vector_type(16))) __bf16 v16bf;
typedef __attribute__((ext_vector_type(8)))  __bf16 v8bf;
typedef __attribute__((ext_vector_type(8)))  float  v8f;

#define BM 64
#define BN 256
#define BK 32
#define LDSK 40  // BK + 8 pad: row pitch = 80B (multiple of 16B, staggers banks)

// ---------------------------------------------------------------- pack f32->bf16
__global__ void pack_bf16_kernel(const float* __restrict__ in,
                                 bf16_t* __restrict__ out, int n) {
  int i = blockIdx.x * blockDim.x + threadIdx.x;
  int stride = gridDim.x * blockDim.x;
  for (; i < n; i += stride) out[i] = (bf16_t)in[i];
}

// ---------------------------------------------------------------- transposed pack:
// in [batch][C][HW] f32  ->  out [batch][HW][C] bf16 ; LDS-tiled 32x32, block (32,8)
__global__ __launch_bounds__(256)
void pack_transpose_kernel(const float* __restrict__ in, bf16_t* __restrict__ out,
                           int C, int HW) {
  __shared__ float tile[32][33];
  int b = blockIdx.z;
  int c0 = blockIdx.y * 32, h0 = blockIdx.x * 32;
  const float* src = in + ((long)b * C + c0) * HW + h0;
#pragma unroll
  for (int i = threadIdx.y; i < 32; i += 8)
    tile[i][threadIdx.x] = src[(long)i * HW + threadIdx.x];
  __syncthreads();
  bf16_t* dst = out + ((long)b * HW + h0) * C + c0;
#pragma unroll
  for (int i = threadIdx.y; i < 32; i += 8)
    dst[(long)i * C + threadIdx.x] = (bf16_t)tile[threadIdx.x][i];
}

// ---------------------------------------------------------------- WMMA GEMM
// C[M,N] = epi( sum_k A[m,k] * B[k,n] )
//   A row-major [M,K]  (lda = k-row stride)
//   B stored    [N,K]  (ldb = k-row stride)  -- i.e. logical B[k,n] = Bmem[n*ldb+k]
// Both operands K-contiguous: staging and fragment loads are all 16B vectors.
// Block: 256 thr (8 waves); block tile 64(M) x 256(N); wave tile 32x64 -> 8 wmma / K-step.
// Register-prefetch software pipeline hides global latency under the wmma burst.
template <bool OUT_BF16>
__global__ __launch_bounds__(256)
void gemm_wmma_kernel(const bf16_t* __restrict__ A, long batchStrideA, int lda,
                      const bf16_t* __restrict__ B, long batchStrideB, int ldb,
                      void* __restrict__ Cout, long batchStrideC, int ldc,
                      const float* __restrict__ biasRow,
                      const float* __restrict__ biasCol,
                      const float* __restrict__ resid, long batchStrideR,
                      float scale, int K) {
  __shared__ __align__(16) bf16_t As[BM][LDSK];  // As[m][k]
  __shared__ __align__(16) bf16_t Bs[BN][LDSK];  // Bs[n][k]

  const int tid   = threadIdx.x;
  const int bm    = blockIdx.y * BM;
  const int bn    = blockIdx.x * BN;
  const int batch = blockIdx.z;

  A += (long)batch * batchStrideA;
  B += (long)batch * batchStrideB;

  const int wave = tid >> 5, lane = tid & 31;
  const int half = lane >> 4, m16 = lane & 15;
  const int wm = (wave & 1) * 32;   // wave M offset in block tile
  const int wn = (wave >> 1) * 64;  // wave N offset in block tile

  // staging addresses (each thread: 8 A elems as one v8bf, 32 B elems as four v8bf)
  const int ar = tid >> 2, ak = (tid & 3) * 8;          // A: row 0..63, k 0,8,16,24
  const bf16_t* Ag = A + (long)(bm + ar) * lda + ak;
  const bf16_t* Bg = B + (long)(bn + tid) * ldb;        // B: n-row = tid

  v8f acc[2][4] = {};
  union U16 { v16bf v; v8bf h[2]; };

  // prologue: prefetch tile 0 into registers
  v8bf pa  = *(const v8bf*)(Ag);
  v8bf pb0 = *(const v8bf*)(Bg);
  v8bf pb1 = *(const v8bf*)(Bg + 8);
  v8bf pb2 = *(const v8bf*)(Bg + 16);
  v8bf pb3 = *(const v8bf*)(Bg + 24);

  for (int k0 = 0; k0 < K; k0 += BK) {
    // commit prefetched tile to LDS
    *(v8bf*)&As[ar][ak] = pa;
    bf16_t* br = &Bs[tid][0];
    *(v8bf*)(br)      = pb0;
    *(v8bf*)(br + 8)  = pb1;
    *(v8bf*)(br + 16) = pb2;
    *(v8bf*)(br + 24) = pb3;
    __syncthreads();

    // issue next tile's global loads early (overlap with wmma burst)
    int kn = k0 + BK;
    if (kn < K) {
      pa  = *(const v8bf*)(Ag + kn);
      pb0 = *(const v8bf*)(Bg + kn);
      pb1 = *(const v8bf*)(Bg + kn + 8);
      pb2 = *(const v8bf*)(Bg + kn + 16);
      pb3 = *(const v8bf*)(Bg + kn + 24);
    }

    // fragments per documented 16-bit WMMA layouts (wave32):
    // A(16x32): lane m = lane&15; elems 0..7 -> K=half*8..+7, elems 8..15 -> K=16+half*8..+7
    // B(32x16): lane n = lane&15; elems 0..7 -> K=half*8..+7, elems 8..15 -> K=16+half*8..+7
    v16bf af[2], bfg[4];
#pragma unroll
    for (int t = 0; t < 2; ++t) {
      const bf16_t* ap = &As[wm + t * 16 + m16][0];
      U16 u;
      u.h[0] = *(const v8bf*)(ap + half * 8);
      u.h[1] = *(const v8bf*)(ap + 16 + half * 8);
      af[t] = u.v;
    }
#pragma unroll
    for (int t = 0; t < 4; ++t) {
      const bf16_t* bp = &Bs[wn + t * 16 + m16][0];
      U16 u;
      u.h[0] = *(const v8bf*)(bp + half * 8);
      u.h[1] = *(const v8bf*)(bp + 16 + half * 8);
      bfg[t] = u.v;
    }
#pragma unroll
    for (int tm = 0; tm < 2; ++tm)
#pragma unroll
      for (int tn = 0; tn < 4; ++tn)
        acc[tm][tn] = __builtin_amdgcn_wmma_f32_16x16x32_bf16(
            false, af[tm], false, bfg[tn], (short)0, acc[tm][tn], false, false);
    __syncthreads();
  }

  // epilogue: C/D f32 layout: VGPR r -> M = r + half*8, N = lane&15
  bf16_t* Cb = (bf16_t*)Cout + (long)batch * batchStrideC;
  float*  Cf = (float*)Cout  + (long)batch * batchStrideC;
  const float* rbase = resid ? resid + (long)batch * batchStrideR : nullptr;
#pragma unroll
  for (int tm = 0; tm < 2; ++tm)
#pragma unroll
    for (int tn = 0; tn < 4; ++tn)
#pragma unroll
      for (int r = 0; r < 8; ++r) {
        int gr = bm + wm + tm * 16 + half * 8 + r;
        int gc = bn + wn + tn * 16 + m16;
        float v = acc[tm][tn][r];
        if (biasRow) v += biasRow[gr];
        if (biasCol) v += biasCol[gc];
        v *= scale;
        long idx = (long)gr * ldc + gc;
        if (rbase) v += rbase[idx];
        if (OUT_BF16) Cb[idx] = (bf16_t)v;
        else          Cf[idx] = v;
      }
}

// ---------------------------------------------------------------- row softmax f32 -> bf16
__global__ __launch_bounds__(256)
void softmax_rows_kernel(const float* __restrict__ scores,
                         bf16_t* __restrict__ attn, int N) {
  __shared__ float red[256];
  long rowBase = ((long)blockIdx.y * gridDim.x + blockIdx.x) * N;
  const float* s = scores + rowBase;
  int tid = threadIdx.x;
  float v[4];
  float m = -3.4e38f;
#pragma unroll
  for (int i = 0; i < 4; ++i) { v[i] = s[tid + i * 256]; m = fmaxf(m, v[i]); }
  red[tid] = m; __syncthreads();
  for (int off = 128; off > 0; off >>= 1) {
    if (tid < off) red[tid] = fmaxf(red[tid], red[tid + off]);
    __syncthreads();
  }
  m = red[0]; __syncthreads();
  float sum = 0.f;
#pragma unroll
  for (int i = 0; i < 4; ++i) { v[i] = __expf(v[i] - m); sum += v[i]; }
  red[tid] = sum; __syncthreads();
  for (int off = 128; off > 0; off >>= 1) {
    if (tid < off) red[tid] += red[tid + off];
    __syncthreads();
  }
  float inv = 1.0f / red[0];
  bf16_t* a = attn + rowBase;
#pragma unroll
  for (int i = 0; i < 4; ++i) a[tid + i * 256] = (bf16_t)(v[i] * inv);
}

// ---------------------------------------------------------------- launch
extern "C" void kernel_launch(void* const* d_in, const int* in_sizes, int n_in,
                              void* d_out, int out_size, void* d_ws, size_t ws_size,
                              hipStream_t stream) {
  (void)in_sizes; (void)n_in; (void)out_size; (void)ws_size;
  const float* x       = (const float*)d_in[0];  // [16,1024,32,32]
  const float* w_theta = (const float*)d_in[1];  // [512,1024]
  const float* b_theta = (const float*)d_in[2];  // [512]
  const float* w_phi   = (const float*)d_in[3];
  const float* b_phi   = (const float*)d_in[4];
  const float* w_g     = (const float*)d_in[5];
  const float* b_g     = (const float*)d_in[6];
  const float* w_out   = (const float*)d_in[7];  // [1024,512]
  const float* b_out   = (const float*)d_in[8];  // [1024]
  float* out = (float*)d_out;                    // [16,1024,32,32]

  const int NB = 16, C = 1024, D = 512, N = 1024;  // N = 32*32

  char* p = (char*)d_ws;
  auto carve = [&](size_t bytes) {
    char* r = p;
    p += (bytes + 255) & ~(size_t)255;
    return r;
  };
  bf16_t* XbT = (bf16_t*)carve((size_t)NB * N * C * 2);  // x^T bf16 [n][hw][c]
  bf16_t* Wt  = (bf16_t*)carve((size_t)D * C * 2);       // [d][c]
  bf16_t* Wp  = (bf16_t*)carve((size_t)D * C * 2);
  bf16_t* Wg  = (bf16_t*)carve((size_t)D * C * 2);
  bf16_t* Wo  = (bf16_t*)carve((size_t)C * D * 2);       // [c][d]
  bf16_t* ThT = (bf16_t*)carve((size_t)NB * N * D * 2);  // theta^T [n][hw][d]
  bf16_t* PhT = (bf16_t*)carve((size_t)NB * N * D * 2);  // phi^T   [n][hw][d]
  bf16_t* G   = (bf16_t*)carve((size_t)NB * D * N * 2);  // g       [n][d][hw]
  float*  Sc  = (float*) carve((size_t)NB * N * N * 4);  // scores  [n][i][j] f32
  bf16_t* At  = (bf16_t*)carve((size_t)NB * N * N * 2);  // attn    [n][i][j]
  bf16_t* Y   = (bf16_t*)carve((size_t)NB * N * D * 2);  // y       [n][hw][d]

  // packing
  dim3 tpg(N / 32, C / 32, NB);
  pack_transpose_kernel<<<tpg, dim3(32, 8), 0, stream>>>(x, XbT, C, N);
  pack_bf16_kernel<<<512, 256, 0, stream>>>(w_theta, Wt, D * C);
  pack_bf16_kernel<<<512, 256, 0, stream>>>(w_phi,   Wp, D * C);
  pack_bf16_kernel<<<512, 256, 0, stream>>>(w_g,     Wg, D * C);
  pack_bf16_kernel<<<512, 256, 0, stream>>>(w_out,   Wo, C * D);

  dim3 blk(256);

  // theta^T [hw,d] = X^T[hw,c] @ Wt^T : A = XbT (lda=C), B = Wt stored [d][c] (ldb=C), col-bias
  dim3 gpT(D / BN, N / BM, NB);  // (2,16,16)
  gemm_wmma_kernel<true><<<gpT, blk, 0, stream>>>(
      XbT, (long)N * C, C, Wt, 0, C, ThT, (long)N * D, D,
      nullptr, b_theta, nullptr, 0, 1.0f, C);
  gemm_wmma_kernel<true><<<gpT, blk, 0, stream>>>(
      XbT, (long)N * C, C, Wp, 0, C, PhT, (long)N * D, D,
      nullptr, b_phi, nullptr, 0, 1.0f, C);

  // g [d,hw] = Wg @ X : A = Wg [d][c] (lda=C), B = XbT stored [hw][c] (ldb=C), row-bias
  dim3 gG(N / BN, D / BM, NB);  // (4,8,16)
  gemm_wmma_kernel<true><<<gG, blk, 0, stream>>>(
      Wg, 0, C, XbT, (long)N * C, C, G, (long)D * N, N,
      b_g, nullptr, nullptr, 0, 1.0f, C);

  // scores [i,j] = theta^T[i,:] . phi^T[j,:] * 1/sqrt(D)
  //   A = ThT (lda=D), B = PhT stored [j][d] (ldb=D), f32 out
  dim3 gS(N / BN, N / BM, NB);  // (4,16,16)
  gemm_wmma_kernel<false><<<gS, blk, 0, stream>>>(
      ThT, (long)N * D, D, PhT, (long)N * D, D, Sc, (long)N * N, N,
      nullptr, nullptr, nullptr, 0, 0.0441941738241592f /* 1/sqrt(512) */, D);

  // softmax rows -> bf16 attn
  softmax_rows_kernel<<<dim3(N, NB), 256, 0, stream>>>(Sc, At, N);

  // y [i,d] = attn[i,:] @ g^T : A = At (lda=N), B = G stored [d][hw] (ldb=N)
  dim3 gY(D / BN, N / BM, NB);  // (2,16,16)
  gemm_wmma_kernel<true><<<gY, blk, 0, stream>>>(
      At, (long)N * N, N, G, (long)D * N, N, Y, (long)N * D, D,
      nullptr, nullptr, nullptr, 0, 1.0f, N);

  // out [c,i] = x + Wo @ y^T : A = Wo [c][d] (lda=D), B = Y stored [i][d] (ldb=D),
  //   row-bias b_out, residual x, f32 out
  dim3 gO(N / BN, C / BM, NB);  // (4,16,16)
  gemm_wmma_kernel<false><<<gO, blk, 0, stream>>>(
      Wo, 0, D, Y, (long)N * D, D, out, (long)C * N, N,
      b_out, nullptr, x, (long)C * N, 1.0f, D);
}
